// decoderHTR_51960514347399
// MI455X (gfx1250) — matrix operand order repaired and checked by reading, not compiled, not run on hardware
//
#include <hip/hip_runtime.h>
#include <math.h>

// Model dims (match reference)
#define NLAY 3
#define NB   8
#define MP   16    // padded row count for activations (one WMMA M-tile)
#define SMEM 128
#define DM   512
#define NH   8
#define HD   64
#define DFFN 2048
#define NV   128
#define LSEQ 32
#define SOS  1
#define EOS  2

typedef __attribute__((ext_vector_type(16))) _Float16 v16h;
typedef __attribute__((ext_vector_type(8)))  _Float16 v8h;
typedef __attribute__((ext_vector_type(8)))  float    v8f;

// ---------------- wave helpers (wave32) ----------------
__device__ __forceinline__ float wave_sum(float v) {
#pragma unroll
  for (int m = 16; m >= 1; m >>= 1) v += __shfl_xor(v, m, 32);
  return v;
}
__device__ __forceinline__ float wave_max(float v) {
#pragma unroll
  for (int m = 16; m >= 1; m >>= 1) v = fmaxf(v, __shfl_xor(v, m, 32));
  return v;
}

// ------- weight convert + transpose: W[K,N] f32 -> Wt[N,K] f16 -------------
__global__ void wcvt_t_kernel(const float* __restrict__ W, _Float16* __restrict__ Wt,
                              int K, int N) {
  int idx = blockIdx.x * 256 + threadIdx.x;
  if (idx < K * N) {
    int k = idx / N;
    int n = idx - k * N;
    Wt[(size_t)n * K + k] = (_Float16)W[idx];
  }
}

// ------- flat f32 -> f16 convert (for `memory` activation) -----------------
__global__ void acvt_kernel(const float* __restrict__ A, _Float16* __restrict__ Ah,
                            int total) {
  int idx = blockIdx.x * 256 + threadIdx.x;
  if (idx < total) Ah[idx] = (_Float16)A[idx];
}

// ------- zero the pad rows (8..15) of step activation buffers --------------
__global__ void zero_pads_kernel(float* __restrict__ x, _Float16* __restrict__ xh,
                                 _Float16* __restrict__ attnh) {
  int i = blockIdx.x * 256 + threadIdx.x;
  if (i < 8 * DM) {
    int off = 8 * DM + i;
    x[off] = 0.f;
    xh[off] = (_Float16)0.f;
    attnh[off] = (_Float16)0.f;
  }
}

// ------- WMMA GEMM: C[M,N] = act(A[Mp,K]f16 @ Wt[N,K]f16 + bias) -----------
// grid = (Mtiles, N/16), block = 32 (one wave per 16x16 tile).
// A rows are padded to a multiple of 16 and always readable.
// C (f32, guarded by Mstore) and/or Ch (f16, all 16 rows) may be null.
__global__ void __launch_bounds__(32)
wmma_gemm_kernel(const _Float16* __restrict__ A, const _Float16* __restrict__ Wt,
                 const float* __restrict__ bias, float* __restrict__ C,
                 _Float16* __restrict__ Ch, int Mstore, int K, int ldc, int relu) {
  const int lane = threadIdx.x;
  const int m0 = blockIdx.x * 16;
  const int n0 = blockIdx.y * 16;
  const int khalf = lane >> 4;

  const _Float16* arow = A + (size_t)(m0 + (lane & 15)) * K + khalf * 16;
  const _Float16* bptr = Wt + (size_t)(n0 + (lane & 15)) * K + khalf * 16;

  v8f acc = {};
#pragma unroll 4
  for (int kb = 0; kb < K; kb += 32) {
    // A fragment: elems 0..15 = K kb+khalf*16..+15
    v8h a0 = *(const v8h*)(arow + kb);
    v8h a1 = *(const v8h*)(arow + kb + 8);
    v16h a;
#pragma unroll
    for (int i = 0; i < 8; ++i) { a[i] = a0[i]; a[8 + i] = a1[i]; }
    // B fragment: lanes 0-15 hold K kb..kb+15 for N = n0+lane; lanes 16-31 K kb+16..+31
    v16h b = *(const v16h*)(bptr + kb);
    __builtin_prefetch(bptr + kb + 128, 0, 1);  // global_prefetch_b8 on weight stream
    acc = __builtin_amdgcn_wmma_f32_16x16x32_f16(false, a, false, b, (short)0, acc,
                                                 false, false);
  }

  const int n = n0 + (lane & 15);
  const float bv = bias[n];
  const int mbase = m0 + (lane >> 4) * 8;
#pragma unroll
  for (int r = 0; r < 8; ++r) {
    int m = mbase + r;
    float v = acc[r] + bv;
    if (relu) v = fmaxf(v, 0.0f);
    if (C && m < Mstore) C[(size_t)m * ldc + n] = v;
    if (Ch) Ch[(size_t)m * ldc + n] = (_Float16)v;
  }
}

// ---------------- embedding + sinusoid PE for current step ----------------
// grid = NB, block = 256. Writes f32 x (residual stream) and f16 xh (GEMM A).
__global__ void embed_kernel(const float* __restrict__ emb, const int* __restrict__ tokens,
                             float* __restrict__ x, _Float16* __restrict__ xh, int t) {
  int b = blockIdx.x;
  int tok = tokens[b * (LSEQ + 1) + t];
  int pos = (t == 0) ? 0 : (t + 1);  // replicate reference pe indexing
  const float scale = 22.62741699796952f;  // sqrt(512)
  for (int d = threadIdx.x; d < DM; d += 256) {
    float div = __expf(-(float)(d & ~1) * (9.210340371976184f / (float)DM));
    float ang = (float)pos * div;
    float pe = (d & 1) ? __cosf(ang) : __sinf(ang);
    float v = emb[tok * DM + d] * scale + pe;
    x[b * DM + d] = v;
    xh[b * DM + d] = (_Float16)v;
  }
}

// ---------------- causal self-attention step (KV cache) -------------------
// grid = NB*NH, block = 32. qkv:[16][1536] (rows 0..7 live); caches:[8][32][512]
__global__ void __launch_bounds__(32)
sa_attn_kernel(const float* __restrict__ qkv, float* __restrict__ kcache,
               float* __restrict__ vcache, _Float16* __restrict__ outh, int t) {
  int bh = blockIdx.x;
  int b = bh >> 3, h = bh & 7;
  int lane = threadIdx.x;

  // append current k,v into cache
  for (int i = lane; i < HD; i += 32) {
    kcache[(b * LSEQ + t) * DM + h * HD + i] = qkv[b * 3 * DM + DM + h * HD + i];
    vcache[(b * LSEQ + t) * DM + h * HD + i] = qkv[b * 3 * DM + 2 * DM + h * HD + i];
  }

  const float* qp = qkv + b * 3 * DM + h * HD;
  float s = -1e30f;
  int j = lane;
  if (j <= t) {
    const float* kj = (j == t) ? (qkv + b * 3 * DM + DM + h * HD)
                               : (kcache + (b * LSEQ + j) * DM + h * HD);
    float acc = 0.f;
#pragma unroll 8
    for (int d = 0; d < HD; ++d) acc += qp[d] * kj[d];
    s = acc * 0.125f;  // 1/sqrt(64)
  }
  float mx = wave_max(s);
  float e = (j <= t) ? __expf(s - mx) : 0.f;
  float sum = wave_sum(e);
  float p = e / sum;

  float o0 = 0.f, o1 = 0.f;
  int d0 = lane * 2;
  for (int jj = 0; jj <= t; ++jj) {
    float pj = __shfl(p, jj, 32);
    const float* vj = (jj == t) ? (qkv + b * 3 * DM + 2 * DM + h * HD)
                                : (vcache + (b * LSEQ + jj) * DM + h * HD);
    o0 += pj * vj[d0];
    o1 += pj * vj[d0 + 1];
  }
  outh[b * DM + h * HD + d0] = (_Float16)o0;
  outh[b * DM + h * HD + d0 + 1] = (_Float16)o1;
}

// ---------------- cross attention over precomputed memory K/V -------------
// grid = NB*NH, block = 32. q:[16][512]; kv:[8][128][1024]
__global__ void __launch_bounds__(32)
ca_attn_kernel(const float* __restrict__ q, const float* __restrict__ kv,
               _Float16* __restrict__ outh) {
  __shared__ float plds[SMEM];
  int bh = blockIdx.x;
  int b = bh >> 3, h = bh & 7;
  int lane = threadIdx.x;
  const float* qp = q + b * DM + h * HD;

  float sv[4];
  float mloc = -1e30f;
#pragma unroll
  for (int jj = 0; jj < 4; ++jj) {
    int j = jj * 32 + lane;
    const float* kj = kv + ((size_t)b * SMEM + j) * (2 * DM) + h * HD;
    float acc = 0.f;
#pragma unroll 8
    for (int d = 0; d < HD; ++d) acc += qp[d] * kj[d];
    sv[jj] = acc * 0.125f;
    mloc = fmaxf(mloc, sv[jj]);
  }
  float mx = wave_max(mloc);
  float es = 0.f;
#pragma unroll
  for (int jj = 0; jj < 4; ++jj) {
    float e = __expf(sv[jj] - mx);
    plds[jj * 32 + lane] = e;
    es += e;
  }
  float sum = wave_sum(es);
  __syncthreads();
  float inv = 1.0f / sum;

  float o0 = 0.f, o1 = 0.f;
  int d0 = lane * 2;
  for (int j = 0; j < SMEM; ++j) {
    float pj = plds[j] * inv;
    const float* vj = kv + ((size_t)b * SMEM + j) * (2 * DM) + DM + h * HD;
    o0 += pj * vj[d0];
    o1 += pj * vj[d0 + 1];
  }
  outh[b * DM + h * HD + d0] = (_Float16)o0;
  outh[b * DM + h * HD + d0 + 1] = (_Float16)o1;
}

// ------- residual + layernorm: x = LN(x + a)*g + bb, also emit f16 --------
// grid = NB, block = 32
__global__ void __launch_bounds__(32)
add_ln_kernel(float* __restrict__ x, _Float16* __restrict__ xh,
              const float* __restrict__ a, const float* __restrict__ g,
              const float* __restrict__ bb) {
  int r = blockIdx.x, lane = threadIdx.x;
  float vals[16];
  float sum = 0.f;
#pragma unroll
  for (int i = 0; i < 16; ++i) {
    int d = lane + i * 32;
    float v = x[r * DM + d] + a[r * DM + d];
    vals[i] = v;
    sum += v;
  }
  float mean = wave_sum(sum) * (1.0f / DM);
  float var = 0.f;
#pragma unroll
  for (int i = 0; i < 16; ++i) {
    float dv = vals[i] - mean;
    var += dv * dv;
  }
  var = wave_sum(var) * (1.0f / DM);
  float inv = rsqrtf(var + 1e-5f);
#pragma unroll
  for (int i = 0; i < 16; ++i) {
    int d = lane + i * 32;
    float o = (vals[i] - mean) * inv * g[d] + bb[d];
    x[r * DM + d] = o;
    xh[r * DM + d] = (_Float16)o;
  }
}

// ---------------- argmax over V=128 logits, append token ------------------
// grid = NB, block = 32. logits row b at logits + b*ldl
__global__ void __launch_bounds__(32)
argmax_kernel(const float* __restrict__ logits, int ldl, int* __restrict__ tokens, int t) {
  int b = blockIdx.x, lane = threadIdx.x;
  const float* lp = logits + (size_t)b * ldl;
  float best = -1e30f;
  int bi = 0;
#pragma unroll
  for (int jj = 0; jj < 4; ++jj) {
    int j = lane + jj * 32;
    float v = lp[j];
    if (v > best) { best = v; bi = j; }
  }
#pragma unroll
  for (int m = 16; m >= 1; m >>= 1) {
    float ov = __shfl_xor(best, m, 32);
    int oi = __shfl_xor(bi, m, 32);
    if (ov > best || (ov == best && oi < bi)) { best = ov; bi = oi; }
  }
  if (lane == 0) tokens[b * (LSEQ + 1) + t + 1] = bi;
}

// ---------------- token init / finalize -----------------------------------
__global__ void init_tokens_kernel(int* tokens) {
  int tid = threadIdx.x;
  if (tid < NB) tokens[tid * (LSEQ + 1)] = SOS;
}

__global__ void finalize_kernel(const int* __restrict__ tokens, int* __restrict__ outp) {
  int b = threadIdx.x;
  if (b < NB) {
    int eos = 0;
    for (int i = 0; i < LSEQ + 1; ++i) {
      if (tokens[b * (LSEQ + 1) + i] == EOS) { eos = i; break; }
    }
    for (int i = 0; i < LSEQ + 1; ++i) {
      int v = tokens[b * (LSEQ + 1) + i];
      if (eos != 0 && i > eos + 1) v = 0;  // PAD
      outp[b * (LSEQ + 1) + i] = v;
    }
  }
}

// ===========================================================================
extern "C" void kernel_launch(void* const* d_in, const int* in_sizes, int n_in,
                              void* d_out, int out_size, void* d_ws, size_t ws_size,
                              hipStream_t stream) {
  (void)in_sizes; (void)n_in; (void)out_size; (void)ws_size;
  const float* memory   = (const float*)d_in[0];
  const float* emb      = (const float*)d_in[1];
  const float* sa_qkv_w = (const float*)d_in[2];
  const float* sa_qkv_b = (const float*)d_in[3];
  const float* sa_out_w = (const float*)d_in[4];
  const float* sa_out_b = (const float*)d_in[5];
  const float* ca_q_w   = (const float*)d_in[6];
  const float* ca_q_b   = (const float*)d_in[7];
  const float* ca_kv_w  = (const float*)d_in[8];
  const float* ca_kv_b  = (const float*)d_in[9];
  const float* ca_out_w = (const float*)d_in[10];
  const float* ca_out_b = (const float*)d_in[11];
  const float* ln1_g = (const float*)d_in[12];
  const float* ln1_b = (const float*)d_in[13];
  const float* ln2_g = (const float*)d_in[14];
  const float* ln2_b = (const float*)d_in[15];
  const float* ln3_g = (const float*)d_in[16];
  const float* ln3_b = (const float*)d_in[17];
  const float* ffn_w1 = (const float*)d_in[18];
  const float* ffn_b1 = (const float*)d_in[19];
  const float* ffn_w2 = (const float*)d_in[20];
  const float* ffn_b2 = (const float*)d_in[21];
  const float* clf_w  = (const float*)d_in[22];
  const float* clf_b  = (const float*)d_in[23];

  // workspace carve-up
  char* ws = (char*)d_ws;
  auto take = [&](size_t bytes) {
    void* p = (void*)ws;
    ws += (bytes + 255) & ~(size_t)255;
    return p;
  };
  _Float16* h_sa_qkv = (_Float16*)take((size_t)NLAY * DM * 3 * DM * 2);
  _Float16* h_sa_out = (_Float16*)take((size_t)NLAY * DM * DM * 2);
  _Float16* h_ca_q   = (_Float16*)take((size_t)NLAY * DM * DM * 2);
  _Float16* h_ca_kv  = (_Float16*)take((size_t)NLAY * DM * 2 * DM * 2);
  _Float16* h_ca_out = (_Float16*)take((size_t)NLAY * DM * DM * 2);
  _Float16* h_ffn1   = (_Float16*)take((size_t)NLAY * DM * DFFN * 2);
  _Float16* h_ffn2   = (_Float16*)take((size_t)NLAY * DFFN * DM * 2);
  _Float16* h_clf    = (_Float16*)take((size_t)DM * NV * 2);
  _Float16* mem_h    = (_Float16*)take((size_t)NB * SMEM * DM * 2);
  float* ca_kv   = (float*)take((size_t)NLAY * NB * SMEM * 2 * DM * 4);
  float* kcache  = (float*)take((size_t)NLAY * NB * LSEQ * DM * 4);
  float* vcache  = (float*)take((size_t)NLAY * NB * LSEQ * DM * 4);
  float*    x     = (float*)take((size_t)MP * DM * 4);
  _Float16* xh    = (_Float16*)take((size_t)MP * DM * 2);
  float*    qkv   = (float*)take((size_t)MP * 3 * DM * 4);  // reused for cross-attn q
  _Float16* attnh = (_Float16*)take((size_t)MP * DM * 2);
  float*    proj  = (float*)take((size_t)MP * DM * 4);
  _Float16* ffh   = (_Float16*)take((size_t)MP * DFFN * 2);
  int*      tokens = (int*)take((size_t)NB * (LSEQ + 1) * 4);

  float* out_logits = (float*)d_out;                                  // [B][L][V]
  int*   out_sampled = (int*)(out_logits + (size_t)NB * LSEQ * NV);   // [B][L+1]

  auto wcvt = [&](const float* W, _Float16* Wt, int K, int N) {
    int total = K * N;
    wcvt_t_kernel<<<(total + 255) / 256, 256, 0, stream>>>(W, Wt, K, N);
  };
  auto gemm = [&](const _Float16* A, const _Float16* Wt, const float* bias,
                  float* C, _Float16* Ch, int Mtiles, int Mstore, int K, int N,
                  int ldc, int relu) {
    dim3 grid(Mtiles, N / 16);
    wmma_gemm_kernel<<<grid, 32, 0, stream>>>(A, Wt, bias, C, Ch, Mstore, K, ldc, relu);
  };

  // ---- one-time prep (per launch) ----
  for (int l = 0; l < NLAY; ++l) {
    wcvt(sa_qkv_w + (size_t)l * DM * 3 * DM, h_sa_qkv + (size_t)l * DM * 3 * DM, DM, 3 * DM);
    wcvt(sa_out_w + (size_t)l * DM * DM,     h_sa_out + (size_t)l * DM * DM,     DM, DM);
    wcvt(ca_q_w   + (size_t)l * DM * DM,     h_ca_q   + (size_t)l * DM * DM,     DM, DM);
    wcvt(ca_kv_w  + (size_t)l * DM * 2 * DM, h_ca_kv  + (size_t)l * DM * 2 * DM, DM, 2 * DM);
    wcvt(ca_out_w + (size_t)l * DM * DM,     h_ca_out + (size_t)l * DM * DM,     DM, DM);
    wcvt(ffn_w1   + (size_t)l * DM * DFFN,   h_ffn1   + (size_t)l * DM * DFFN,   DM, DFFN);
    wcvt(ffn_w2   + (size_t)l * DFFN * DM,   h_ffn2   + (size_t)l * DFFN * DM,   DFFN, DM);
  }
  wcvt(clf_w, h_clf, DM, NV);
  acvt_kernel<<<(NB * SMEM * DM + 255) / 256, 256, 0, stream>>>(memory, mem_h,
                                                                NB * SMEM * DM);
  zero_pads_kernel<<<(8 * DM + 255) / 256, 256, 0, stream>>>(x, xh, attnh);

  // cross-attn K/V precompute: [B*S_MEM, D] @ [D, 2D] + b  (bulk WMMA GEMM)
  for (int l = 0; l < NLAY; ++l) {
    gemm(mem_h, h_ca_kv + (size_t)l * DM * 2 * DM, ca_kv_b + (size_t)l * 2 * DM,
         ca_kv + (size_t)l * NB * SMEM * 2 * DM, nullptr,
         (NB * SMEM) / 16, NB * SMEM, DM, 2 * DM, 2 * DM, 0);
  }

  init_tokens_kernel<<<1, 32, 0, stream>>>(tokens);

  // ---- 32 sequential decode steps ----
  for (int t = 0; t < LSEQ; ++t) {
    embed_kernel<<<NB, 256, 0, stream>>>(emb, tokens, x, xh, t);

    for (int l = 0; l < NLAY; ++l) {
      float* kc = kcache + (size_t)l * NB * LSEQ * DM;
      float* vc = vcache + (size_t)l * NB * LSEQ * DM;
      const float* kvl = ca_kv + (size_t)l * NB * SMEM * 2 * DM;

      // self-attention
      gemm(xh, h_sa_qkv + (size_t)l * DM * 3 * DM, sa_qkv_b + (size_t)l * 3 * DM,
           qkv, nullptr, 1, MP, DM, 3 * DM, 3 * DM, 0);
      sa_attn_kernel<<<NB * NH, 32, 0, stream>>>(qkv, kc, vc, attnh, t);
      gemm(attnh, h_sa_out + (size_t)l * DM * DM, sa_out_b + (size_t)l * DM,
           proj, nullptr, 1, MP, DM, DM, DM, 0);
      add_ln_kernel<<<NB, 32, 0, stream>>>(x, xh, proj,
                                           ln1_g + (size_t)l * DM, ln1_b + (size_t)l * DM);

      // cross-attention
      gemm(xh, h_ca_q + (size_t)l * DM * DM, ca_q_b + (size_t)l * DM,
           qkv, nullptr, 1, MP, DM, DM, DM, 0);
      ca_attn_kernel<<<NB * NH, 32, 0, stream>>>(qkv, kvl, attnh);
      gemm(attnh, h_ca_out + (size_t)l * DM * DM, ca_out_b + (size_t)l * DM,
           proj, nullptr, 1, MP, DM, DM, DM, 0);
      add_ln_kernel<<<NB, 32, 0, stream>>>(x, xh, proj,
                                           ln2_g + (size_t)l * DM, ln2_b + (size_t)l * DM);

      // FFN (hidden kept in f16 only: directly consumable as next GEMM A)
      gemm(xh, h_ffn1 + (size_t)l * DM * DFFN, ffn_b1 + (size_t)l * DFFN,
           nullptr, ffh, 1, MP, DM, DFFN, DFFN, 1);
      gemm(ffh, h_ffn2 + (size_t)l * DFFN * DM, ffn_b2 + (size_t)l * DM,
           proj, nullptr, 1, MP, DFFN, DM, DM, 0);
      add_ln_kernel<<<NB, 32, 0, stream>>>(x, xh, proj,
                                           ln3_g + (size_t)l * DM, ln3_b + (size_t)l * DM);
    }

    // classifier logits -> d_out[b][t][:] via ldc = L*V (store guarded to 8 rows)
    gemm(xh, h_clf, clf_b, out_logits + (size_t)t * NV, nullptr,
         1, NB, DM, NV, LSEQ * NV, 0);
    argmax_kernel<<<NB, 32, 0, stream>>>(out_logits + (size_t)t * NV, LSEQ * NV, tokens, t);
  }

  finalize_kernel<<<1, 32, 0, stream>>>(tokens, out_sampled);
}
